// GatedAttentionPooling_27135603376461
// MI455X (gfx1250) — matrix operand order repaired and checked by reading, not compiled, not run on hardware
//
#include <hip/hip_runtime.h>
#include <hip/hip_bf16.h>
#include <math.h>

#define N_ROWS 262144
#define F_DIM  512
#define AD     256
#define B_SEG  64

typedef __attribute__((ext_vector_type(16))) __bf16 v16bf;
typedef __attribute__((ext_vector_type(8)))  float  v8f;

__device__ __forceinline__ unsigned short f2bf(float f) {
    unsigned u = __float_as_uint(f);
    unsigned r = u + 0x7fffu + ((u >> 16) & 1u);   // round-to-nearest-even
    return (unsigned short)(r >> 16);
}
__device__ __forceinline__ unsigned f2ord(float f) {
    unsigned b = __float_as_uint(f);
    return (b & 0x80000000u) ? ~b : (b | 0x80000000u);
}
__device__ __forceinline__ float ord2f(unsigned u) {
    return (u & 0x80000000u) ? __uint_as_float(u & 0x7fffffffu) : __uint_as_float(~u);
}

// ---------------- Kernel 0: prep (weight bf16 conversion + scratch init) ----
__global__ void prep_kernel(const float* __restrict__ Vw, const float* __restrict__ Uw,
                            unsigned short* __restrict__ Vwb, unsigned short* __restrict__ Uwb,
                            unsigned* __restrict__ gmax, float* __restrict__ gsum,
                            float* __restrict__ outp) {
    int tid = blockIdx.x * blockDim.x + threadIdx.x;
    const int WN = AD * F_DIM; // 131072
    if (tid < WN)            Vwb[tid]      = f2bf(Vw[tid]);
    else if (tid < 2 * WN)   Uwb[tid - WN] = f2bf(Uw[tid - WN]);
    if (tid < B_SEG) { gmax[tid] = 0u; gsum[tid] = 0.0f; }
    if (tid < B_SEG * F_DIM) outp[tid] = 0.0f;
}

// ---------------- Kernel 1: fused dual-GEMM + gate -> per-row logits --------
__global__ __launch_bounds__(128)
void logits_kernel(const float* __restrict__ x,
                   const unsigned short* __restrict__ Vwb,
                   const unsigned short* __restrict__ Uwb,
                   const float* __restrict__ Vb, const float* __restrict__ Ub,
                   const float* __restrict__ ww, const float* __restrict__ wb,
                   float* __restrict__ A) {
    extern __shared__ unsigned short lds[];
    const int LP = F_DIM + 8;                 // padded LDS row stride (bank spread)
    const int tid = threadIdx.x;
    const int blk = blockIdx.x;

    // --- stage 64 rows of x into LDS as bf16 (coalesced float4 reads) ---
    const float4* x4 = (const float4*)(x + (size_t)blk * 64 * F_DIM);
    for (int i = tid; i < 64 * F_DIM / 4; i += 128) {
        float4 v = x4[i];
        int row = i >> 7;                     // 128 float4 per row
        int col = (i & 127) << 2;
        unsigned short* p = &lds[row * LP + col];
        p[0] = f2bf(v.x); p[1] = f2bf(v.y); p[2] = f2bf(v.z); p[3] = f2bf(v.w);
    }
    __syncthreads();

    const int lane = tid & 31;
    const int wv   = tid >> 5;
    const int hi   = lane >> 4;               // 0 for lanes 0-15, 1 for 16-31
    const int lrow = wv * 16 + (lane & 15);

    // --- load all 16 A-fragments (K = 512) into registers, ISA layout ---
    // lane<16: K e0..7 -> k+0..7, e8..15 -> k+16..23 ; lane>=16: +8 shift
    v16bf af[16];
    const unsigned short* rp = &lds[lrow * LP];
    #pragma unroll
    for (int kb = 0; kb < 16; ++kb) {
        int b0 = kb * 32 + hi * 8;
        union { v16bf v; uint4 q[2]; } ua;
        ua.q[0] = *(const uint4*)(rp + b0);
        ua.q[1] = *(const uint4*)(rp + b0 + 16);
        af[kb] = ua.v;
    }

    float rowsum[8] = {0,0,0,0,0,0,0,0};
    const int m0 = blk * 64 + wv * 16;

    for (int ct = 0; ct < 16; ++ct) {         // 16 column tiles -> 256 attn dims
        const int col = ct * 16 + (lane & 15);
        // B layout: same as A: e0..7 -> k+8*hi+0..7, e8..15 -> k+16+8*hi+0..7
        const unsigned short* vp = Vwb + (size_t)col * F_DIM + hi * 8;
        const unsigned short* up = Uwb + (size_t)col * F_DIM + hi * 8;
        v8f accV = {}; v8f accU = {};
        #pragma unroll
        for (int kb = 0; kb < 16; ++kb) {
            union { v16bf v; uint4 q[2]; } bV, bU;
            bV.q[0] = *(const uint4*)(vp + kb * 32);
            bV.q[1] = *(const uint4*)(vp + kb * 32 + 16);
            bU.q[0] = *(const uint4*)(up + kb * 32);
            bU.q[1] = *(const uint4*)(up + kb * 32 + 16);
            accV = __builtin_amdgcn_wmma_f32_16x16x32_bf16(
                       false, af[kb], false, bV.v, (short)0, accV, false, false);
            accU = __builtin_amdgcn_wmma_f32_16x16x32_bf16(
                       false, af[kb], false, bU.v, (short)0, accU, false, false);
        }
        // epilogue: C layout row = r + 8*hi, col = lane%16 (+16*ct)
        const float vb = Vb[col], ub = Ub[col], wc = ww[col];
        #pragma unroll
        for (int r = 0; r < 8; ++r) {
            float t = tanhf(accV[r] + vb);
            float s = 1.0f / (1.0f + __expf(-(accU[r] + ub)));
            rowsum[r] += t * s * wc;
        }
    }
    // reduce the 16 columns across lanes (xor stays within 16-lane halves)
    #pragma unroll
    for (int r = 0; r < 8; ++r) {
        float v = rowsum[r];
        v += __shfl_xor(v, 1, 32);
        v += __shfl_xor(v, 2, 32);
        v += __shfl_xor(v, 4, 32);
        v += __shfl_xor(v, 8, 32);
        rowsum[r] = v;
    }
    const float wbv = wb[0];
    if (lane == 0) {
        #pragma unroll
        for (int r = 0; r < 8; ++r) A[m0 + r] = rowsum[r] + wbv;        // rows 0-7
    } else if (lane == 16) {
        #pragma unroll
        for (int r = 0; r < 8; ++r) A[m0 + 8 + r] = rowsum[r] + wbv;    // rows 8-15
    }
}

// ---------------- Kernel 2a: per-segment max (order-preserving uint) --------
__global__ void segmax_kernel(const float* __restrict__ A, const int* __restrict__ seg,
                              unsigned* __restrict__ gmax) {
    __shared__ unsigned smax[B_SEG];
    int tid = threadIdx.x;
    if (tid < B_SEG) smax[tid] = 0u;
    __syncthreads();
    for (int i = blockIdx.x * blockDim.x + tid; i < N_ROWS; i += gridDim.x * blockDim.x)
        atomicMax(&smax[seg[i]], f2ord(A[i]));
    __syncthreads();
    if (tid < B_SEG) atomicMax(&gmax[tid], smax[tid]);
}

// ---------------- Kernel 2b: per-segment exp-sum ----------------------------
__global__ void segsum_kernel(const float* __restrict__ A, const int* __restrict__ seg,
                              const unsigned* __restrict__ gmax, float* __restrict__ gsum) {
    __shared__ float ssum[B_SEG];
    int tid = threadIdx.x;
    if (tid < B_SEG) ssum[tid] = 0.0f;
    __syncthreads();
    for (int i = blockIdx.x * blockDim.x + tid; i < N_ROWS; i += gridDim.x * blockDim.x) {
        int s = seg[i];
        atomicAdd(&ssum[s], __expf(A[i] - ord2f(gmax[s])));
    }
    __syncthreads();
    if (tid < B_SEG) atomicAdd(&gsum[tid], ssum[tid]);
}

// ---------------- Kernel 3: weighted segment pooling ------------------------
__global__ __launch_bounds__(256)
void pool_kernel(const float* __restrict__ x, const int* __restrict__ seg,
                 const float* __restrict__ A, const unsigned* __restrict__ gmax,
                 const float* __restrict__ gsum, float* __restrict__ out) {
    __shared__ float w_s[64];
    __shared__ int   seg_s[64];
    int tid  = threadIdx.x;
    int base = blockIdx.x * 64;
    if (tid < 64) {
        int n = base + tid;
        int s = seg[n];
        w_s[tid]   = __expf(A[n] - ord2f(gmax[s])) / (gsum[s] + 1e-9f);
        seg_s[tid] = s;
    }
    __syncthreads();
    int f0 = tid, f1 = tid + 256;
    float a0 = 0.0f, a1 = 0.0f;
    int cur = seg_s[0];
    for (int r = 0; r < 64; ++r) {
        int s = seg_s[r];
        if (s != cur) {   // sorted segments -> rare flush
            atomicAdd(&out[cur * F_DIM + f0], a0);
            atomicAdd(&out[cur * F_DIM + f1], a1);
            a0 = 0.0f; a1 = 0.0f; cur = s;
        }
        const float* xr = x + (size_t)(base + r) * F_DIM;
        float wv = w_s[r];
        a0 += xr[f0] * wv;
        a1 += xr[f1] * wv;
    }
    atomicAdd(&out[cur * F_DIM + f0], a0);
    atomicAdd(&out[cur * F_DIM + f1], a1);
}

extern "C" void kernel_launch(void* const* d_in, const int* in_sizes, int n_in,
                              void* d_out, int out_size, void* d_ws, size_t ws_size,
                              hipStream_t stream) {
    const float* x  = (const float*)d_in[0];
    const int*   sg = (const int*)d_in[1];
    const float* Vw = (const float*)d_in[3];
    const float* Vb = (const float*)d_in[4];
    const float* Uw = (const float*)d_in[5];
    const float* Ub = (const float*)d_in[6];
    const float* ww = (const float*)d_in[7];
    const float* wb = (const float*)d_in[8];
    float* out = (float*)d_out;

    char* ws = (char*)d_ws;
    unsigned short* Vwb = (unsigned short*)ws;
    unsigned short* Uwb = Vwb + AD * F_DIM;
    float*    Alog = (float*)(ws + (size_t)2 * AD * F_DIM * sizeof(unsigned short));
    unsigned* gmax = (unsigned*)(Alog + N_ROWS);
    float*    gsum = (float*)(gmax + B_SEG);

    prep_kernel<<<1024, 256, 0, stream>>>(Vw, Uw, Vwb, Uwb, gmax, gsum, out);

    size_t ldsBytes = (size_t)64 * (F_DIM + 8) * sizeof(unsigned short);
    logits_kernel<<<N_ROWS / 64, 128, ldsBytes, stream>>>(x, Vwb, Uwb, Vb, Ub, ww, wb, Alog);

    segmax_kernel<<<512, 256, 0, stream>>>(Alog, sg, gmax);
    segsum_kernel<<<512, 256, 0, stream>>>(Alog, sg, gmax, gsum);
    pool_kernel<<<N_ROWS / 64, 256, 0, stream>>>(x, sg, Alog, gmax, gsum, out);
}
